// MultiHeadAttention_69939247448470
// MI455X (gfx1250) — matrix operand order repaired and checked by reading, not compiled, not run on hardware
//
#include <hip/hip_runtime.h>
#include <hip/hip_bf16.h>

#define BB 2
#define SS 2048
#define DD 1024
#define HH 16
#define DKK 64
#define MM (BB * SS) /* 4096 */

typedef __bf16 bf16_t;
typedef __attribute__((ext_vector_type(16))) __bf16 v16bf;
typedef __attribute__((ext_vector_type(8))) float v8f;

// ---------------- WMMA helpers (CDNA5 16x16x32 bf16, f32 accumulate) ----------------

static __device__ __forceinline__ v8f wmma_bf16(v16bf a, v16bf b, v8f c) {
  return __builtin_amdgcn_wmma_f32_16x16x32_bf16(false, a, false, b, (short)0, c,
                                                 false, false);
}

// A fragment: 16x32 bf16, source row-major [16][>=32] in LDS.
static __device__ __forceinline__ v16bf lds_load_A(const bf16_t* p, int stride) {
  const int lane = threadIdx.x & 31;
  const bf16_t* r = p + (lane & 15) * stride + ((lane >> 4) << 4);
  v16bf a;
#pragma unroll
  for (int i = 0; i < 16; ++i) {
    a[i] = r[i];
  }
  return a;
}

// B fragment: 32x16 bf16, source row-major [K][N] with given stride.
static __device__ __forceinline__ v16bf lds_load_B(const bf16_t* p, int stride) {
  const int lane = threadIdx.x & 31;
  const bf16_t* c = p + (lane & 15) + (((lane >> 4) << 4) * stride);
  v16bf b;
#pragma unroll
  for (int i = 0; i < 16; ++i) b[i] = c[i * stride];
  return b;
}

// B fragment from a transposed source: B(k,n) = p[n*stride + k].
static __device__ __forceinline__ v16bf lds_load_Bt(const bf16_t* p, int stride) {
  const int lane = threadIdx.x & 31;
  const bf16_t* c = p + (lane & 15) * stride + ((lane >> 4) << 4);
  v16bf b;
#pragma unroll
  for (int i = 0; i < 16; ++i) b[i] = c[i];
  return b;
}

// ---------------- CDNA5 async global->LDS copy (ASYNCcnt-tracked) ----------------
// Copies a 64x64 bf16 tile (row stride DD in global, 72 in LDS -> 144B = 16B aligned).
// 128 threads x 4 issues x 16B = 8KB.

static __device__ __forceinline__ void async_tile64(const bf16_t* gbase, bf16_t* lds,
                                                    int tid) {
#pragma unroll
  for (int it = 0; it < 4; ++it) {
    const int i = tid + it * 128;        // 512 chunks of 8 bf16 (16B)
    const int row = i >> 3;
    const int col = (i & 7) << 3;
    const bf16_t* g = gbase + (size_t)row * DD + col;
    const unsigned loff = (unsigned)(uintptr_t)(lds + row * 72 + col);
    asm volatile("global_load_async_to_lds_b128 %0, %1, off"
                 :
                 : "v"(loff), "v"((unsigned long long)(uintptr_t)g)
                 : "memory");
  }
}

static __device__ __forceinline__ void wait_async0() {
  asm volatile("s_wait_asynccnt 0x0" ::: "memory");
}

// ---------------- GEMM: C[M,N] = A[M,K] * W[K,N] (+bias), bf16 WMMA ----------------
// 128x64 block tile, 256 threads = 8 waves (4x2 wave grid, 32x32 per wave), K-step 32.

template <bool A_F32, bool OUT_F32>
__global__ __launch_bounds__(256) void gemm128(const void* __restrict__ Ap,
                                               const float* __restrict__ W,
                                               void* __restrict__ Cp,
                                               const float* __restrict__ bias,
                                               int Kdim, int Ndim) {
  const int SA = 40;  // 32 + 8 pad
  const int SB = 72;  // 64 + 8 pad
  __shared__ bf16_t As[128 * 40];
  __shared__ bf16_t Bs[32 * 72];

  const int tid = threadIdx.x;
  const int wave = tid >> 5;
  const int wm = wave >> 1, wn = wave & 1;  // 4 x 2
  const int lane = tid & 31;
  const int m0 = blockIdx.y * 128;
  const int n0 = blockIdx.x * 64;

  v8f acc[2][2];
#pragma unroll
  for (int mi = 0; mi < 2; ++mi)
#pragma unroll
    for (int ni = 0; ni < 2; ++ni)
#pragma unroll
      for (int r = 0; r < 8; ++r) acc[mi][ni][r] = 0.0f;

  for (int k0 = 0; k0 < Kdim; k0 += 32) {
    __syncthreads();
    // ---- stage A tile 128x32 (convert f32->bf16 if needed) ----
    if constexpr (A_F32) {
      const float* A = (const float*)Ap;
#pragma unroll
      for (int it = 0; it < 4; ++it) {
        int i = tid + it * 256;  // 1024 float4s
        int row = i >> 3;
        int col = (i & 7) << 2;
        float4 v = *(const float4*)(A + (size_t)(m0 + row) * Kdim + k0 + col);
        bf16_t* d = &As[row * SA + col];
        d[0] = (bf16_t)v.x; d[1] = (bf16_t)v.y; d[2] = (bf16_t)v.z; d[3] = (bf16_t)v.w;
      }
    } else {
      const bf16_t* A = (const bf16_t*)Ap;
#pragma unroll
      for (int it = 0; it < 4; ++it) {
        int i = tid + it * 256;
        int row = i >> 3;
        int col = (i & 7) << 2;
        const bf16_t* s = A + (size_t)(m0 + row) * Kdim + k0 + col;
        bf16_t* d = &As[row * SA + col];
        d[0] = s[0]; d[1] = s[1]; d[2] = s[2]; d[3] = s[3];
      }
    }
    // ---- stage B tile 32x64 from f32 weights ----
#pragma unroll
    for (int it = 0; it < 2; ++it) {
      int i = tid + it * 256;  // 512 float4s
      int row = i >> 4;
      int col = (i & 15) << 2;
      float4 v = *(const float4*)(W + (size_t)(k0 + row) * Ndim + n0 + col);
      bf16_t* d = &Bs[row * SB + col];
      d[0] = (bf16_t)v.x; d[1] = (bf16_t)v.y; d[2] = (bf16_t)v.z; d[3] = (bf16_t)v.w;
    }
    __syncthreads();

    v16bf af0 = lds_load_A(&As[(wm * 32) * SA], SA);
    v16bf af1 = lds_load_A(&As[(wm * 32 + 16) * SA], SA);
    v16bf bf0 = lds_load_B(&Bs[wn * 32], SB);
    v16bf bf1 = lds_load_B(&Bs[wn * 32 + 16], SB);

    acc[0][0] = wmma_bf16(af0, bf0, acc[0][0]);
    acc[0][1] = wmma_bf16(af0, bf1, acc[0][1]);
    acc[1][0] = wmma_bf16(af1, bf0, acc[1][0]);
    acc[1][1] = wmma_bf16(af1, bf1, acc[1][1]);
  }

  const int n = lane & 15, hi = lane >> 4;
#pragma unroll
  for (int mi = 0; mi < 2; ++mi)
#pragma unroll
    for (int ni = 0; ni < 2; ++ni) {
      const int col = n0 + wn * 32 + ni * 16 + n;
#pragma unroll
      for (int r = 0; r < 8; ++r) {
        const int row = m0 + wm * 32 + mi * 16 + r + hi * 8;
        if constexpr (OUT_F32) {
          ((float*)Cp)[(size_t)row * Ndim + col] = acc[mi][ni][r] + bias[col];
        } else {
          ((bf16_t*)Cp)[(size_t)row * Ndim + col] = (bf16_t)acc[mi][ni][r];
        }
      }
    }
}

// ---------------- Attention: per (b,h,64 query rows); 4 waves, 16 rows/wave ----------------
// Pass A: streaming row max / sum-of-exp.  Pass B: write softmax weights + P*V.
// K/V tiles double-buffered via global_load_async_to_lds_b128 (ASYNCcnt).

__global__ __launch_bounds__(128) void attn_kernel(const bf16_t* __restrict__ Qb,
                                                   const bf16_t* __restrict__ Kb,
                                                   const bf16_t* __restrict__ Vb,
                                                   float* __restrict__ attn_out,
                                                   bf16_t* __restrict__ Xb) {
  const int ST = 72;  // 64 + 8 pad (144B rows -> 16B aligned)
  __shared__ bf16_t Qs[64 * 72];
  __shared__ bf16_t Ks[2][64 * 72];
  __shared__ bf16_t Vs[2][64 * 72];
  __shared__ bf16_t Ps[4 * 16 * 72];

  const int tid = threadIdx.x;
  const int wave = tid >> 5;
  const int lane = tid & 31;
  const int qt = blockIdx.x;
  const int h = blockIdx.y;
  const int b = blockIdx.z;
  const int qbase = qt * 64;
  const int NT = SS / 64;  // 32 key tiles

  const size_t headoff = (size_t)h * DKK;
  const bf16_t* Q = Qb + (size_t)b * SS * DD + headoff;
  const bf16_t* K = Kb + (size_t)b * SS * DD + headoff;
  const bf16_t* V = Vb + (size_t)b * SS * DD + headoff;

  // async-stage Q tile and first K tile
  async_tile64(Q + (size_t)qbase * DD, Qs, tid);
  async_tile64(K, Ks[0], tid);
  wait_async0();
  __syncthreads();

  const v16bf aq0 = lds_load_A(&Qs[(wave * 16) * ST], ST);       // dk 0..31
  const v16bf aq1 = lds_load_A(&Qs[(wave * 16) * ST + 32], ST);  // dk 32..63

  const float scale = 0.125f;  // 1/sqrt(64)
  float m[8], l[8];
#pragma unroll
  for (int r = 0; r < 8; ++r) { m[r] = -1e30f; l[r] = 0.0f; }

  // ---------------- Pass A ----------------
  for (int kt = 0; kt < NT; ++kt) {
    const int cur = kt & 1;
    if (kt + 1 < NT) async_tile64(K + (size_t)(kt + 1) * 64 * DD, Ks[cur ^ 1], tid);

    const bf16_t* Kc = Ks[cur];
    v8f sc[4];
#pragma unroll
    for (int nt = 0; nt < 4; ++nt) {
      v8f s0 = {0.f, 0.f, 0.f, 0.f, 0.f, 0.f, 0.f, 0.f};
      s0 = wmma_bf16(aq0, lds_load_Bt(&Kc[(nt * 16) * ST], ST), s0);
      s0 = wmma_bf16(aq1, lds_load_Bt(&Kc[(nt * 16) * ST + 32], ST), s0);
      sc[nt] = s0;
    }
#pragma unroll
    for (int r = 0; r < 8; ++r) {
      float pm = fmaxf(fmaxf(sc[0][r], sc[1][r]), fmaxf(sc[2][r], sc[3][r])) * scale;
#pragma unroll
      for (int msk = 1; msk < 16; msk <<= 1) pm = fmaxf(pm, __shfl_xor(pm, msk));
      const float nm = fmaxf(m[r], pm);
      float ps = 0.0f;
#pragma unroll
      for (int nt = 0; nt < 4; ++nt) ps += __expf(sc[nt][r] * scale - nm);
#pragma unroll
      for (int msk = 1; msk < 16; msk <<= 1) ps += __shfl_xor(ps, msk);
      l[r] = l[r] * __expf(m[r] - nm) + ps;
      m[r] = nm;
    }
    wait_async0();   // next tile landed
    __syncthreads(); // everyone done reading current tile
  }
  float linv[8];
#pragma unroll
  for (int r = 0; r < 8; ++r) linv[r] = 1.0f / l[r];

  // ---------------- Pass B ----------------
  v8f o[4];
#pragma unroll
  for (int n2 = 0; n2 < 4; ++n2)
#pragma unroll
    for (int r = 0; r < 8; ++r) o[n2][r] = 0.0f;

  bf16_t* Pw = &Ps[wave * 16 * ST];
  const int n = lane & 15, hi = lane >> 4;

  async_tile64(K, Ks[0], tid);
  async_tile64(V, Vs[0], tid);
  wait_async0();
  __syncthreads();

  for (int kt = 0; kt < NT; ++kt) {
    const int cur = kt & 1;
    if (kt + 1 < NT) {
      async_tile64(K + (size_t)(kt + 1) * 64 * DD, Ks[cur ^ 1], tid);
      async_tile64(V + (size_t)(kt + 1) * 64 * DD, Vs[cur ^ 1], tid);
    }
    const bf16_t* Kc = Ks[cur];
    const bf16_t* Vc = Vs[cur];

#pragma unroll
    for (int nt = 0; nt < 4; ++nt) {
      v8f s0 = {0.f, 0.f, 0.f, 0.f, 0.f, 0.f, 0.f, 0.f};
      s0 = wmma_bf16(aq0, lds_load_Bt(&Kc[(nt * 16) * ST], ST), s0);
      s0 = wmma_bf16(aq1, lds_load_Bt(&Kc[(nt * 16) * ST + 32], ST), s0);
      const int key = kt * 64 + nt * 16 + n;
#pragma unroll
      for (int r = 0; r < 8; ++r) {
        const float w = __expf(s0[r] * scale - m[r]) * linv[r];
        const int qrow = qbase + wave * 16 + r + hi * 8;
        __builtin_nontemporal_store(
            w, &attn_out[(((size_t)b * HH + h) * SS + qrow) * SS + key]);
        Pw[(r + hi * 8) * ST + nt * 16 + n] = (bf16_t)w;
      }
    }
    __syncthreads();

    const v16bf ap0 = lds_load_A(Pw, ST);       // keys 0..31 of tile
    const v16bf ap1 = lds_load_A(Pw + 32, ST);  // keys 32..63
#pragma unroll
    for (int n2 = 0; n2 < 4; ++n2) {
      o[n2] = wmma_bf16(ap0, lds_load_B(&Vc[n2 * 16], ST), o[n2]);
      o[n2] = wmma_bf16(ap1, lds_load_B(&Vc[32 * ST + n2 * 16], ST), o[n2]);
    }
    wait_async0();
    __syncthreads();
  }

  // write X (bf16) for the output projection
#pragma unroll
  for (int n2 = 0; n2 < 4; ++n2)
#pragma unroll
    for (int r = 0; r < 8; ++r) {
      const int qrow = qbase + wave * 16 + r + hi * 8;
      Xb[((size_t)b * SS + qrow) * DD + h * DKK + n2 * 16 + n] = (bf16_t)o[n2][r];
    }
}

// ---------------- host ----------------

extern "C" void kernel_launch(void* const* d_in, const int* in_sizes, int n_in,
                              void* d_out, int out_size, void* d_ws, size_t ws_size,
                              hipStream_t stream) {
  (void)in_sizes; (void)n_in; (void)out_size; (void)ws_size;

  const float* q = (const float*)d_in[0];
  const float* k = (const float*)d_in[1];
  const float* v = (const float*)d_in[2];
  const float* w_q = (const float*)d_in[3];
  const float* w_k = (const float*)d_in[4];
  const float* w_v = (const float*)d_in[5];
  const float* w_o = (const float*)d_in[6];
  const float* b_o = (const float*)d_in[7];

  float* out = (float*)d_out;           // [B,S,D]
  float* attn = out + (size_t)MM * DD;  // [B,H,S,S]

  bf16_t* Qb = (bf16_t*)d_ws;
  bf16_t* Kb = Qb + (size_t)MM * DD;
  bf16_t* Vb = Kb + (size_t)MM * DD;
  bf16_t* Xb = Vb + (size_t)MM * DD;  // 32 MB total bf16 scratch

  dim3 ggrid(DD / 64, MM / 128);  // (16, 32)
  gemm128<true, false><<<ggrid, 256, 0, stream>>>((const void*)q, w_q, (void*)Qb,
                                                  nullptr, DD, DD);
  gemm128<true, false><<<ggrid, 256, 0, stream>>>((const void*)k, w_k, (void*)Kb,
                                                  nullptr, DD, DD);
  gemm128<true, false><<<ggrid, 256, 0, stream>>>((const void*)v, w_v, (void*)Vb,
                                                  nullptr, DD, DD);

  dim3 agrid(SS / 64, HH, BB);  // (32, 16, 2)
  attn_kernel<<<agrid, 128, 0, stream>>>(Qb, Kb, Vb, attn, Xb);

  gemm128<false, true><<<ggrid, 256, 0, stream>>>((const void*)Xb, w_o, (void*)out,
                                                  b_o, DD, DD);
}
